// Attention_53790170415368
// MI455X (gfx1250) — matrix operand order repaired and checked by reading, not compiled, not run on hardware
//
#include <hip/hip_runtime.h>
#include <hip/hip_bf16.h>

// ---------------------------------------------------------------------------
// Problem constants (from the reference)
// ---------------------------------------------------------------------------
#define DIMN   2048
#define HEADS  16
#define HD     128
#define SEQ    2560
#define L_CAP  256
#define NQ     2304          // SEQ - L_CAP : rows entering both SDPA passes

typedef unsigned short u16;
typedef __attribute__((ext_vector_type(16))) __bf16 v16bf;
typedef __attribute__((ext_vector_type(8)))  __bf16 v8bf;
typedef __attribute__((ext_vector_type(8)))  float  v8f;
typedef int v4i __attribute__((vector_size(16)));   // matches async-LDS builtin proto

#define AS_GLOBAL __attribute__((address_space(1)))
#define AS_LOCAL  __attribute__((address_space(3)))

// f32 -> bf16, round-to-nearest-even (bit trick; inputs are finite gaussians)
__device__ __forceinline__ u16 f2bf_bits(float f) {
  unsigned u = __float_as_uint(f);
  unsigned r = u + 0x7FFFu + ((u >> 16) & 1u);
  return (u16)(r >> 16);
}

// Build a v16bf A/B fragment from two 16-byte (8 x bf16) chunks.
// Matches the 16-bit A-matrix VGPR striping: per lane, K {0..7} then K {16..23}
// (lanes >= 16 shift the K base by 8 via the caller's `koff`).
__device__ __forceinline__ v16bf ld_frag16(const u16* lo, const u16* hi) {
  v8bf a = *reinterpret_cast<const v8bf*>(lo);
  v8bf b = *reinterpret_cast<const v8bf*>(hi);
  v16bf r;
#pragma unroll
  for (int i = 0; i < 8; ++i) { r[i] = a[i]; r[i + 8] = b[i]; }
  return r;
}

// D = A(16x32 bf16) * B(32x16 bf16) + C(16x16 f32)  -> v_wmma_f32_16x16x32_bf16
__device__ __forceinline__ v8f wmma_bf16(v16bf a, v16bf b, v8f c) {
  return __builtin_amdgcn_wmma_f32_16x16x32_bf16(
      /*neg_a=*/false, a, /*neg_b=*/false, b,
      /*c_mod=*/(short)0, c, /*reuse_a=*/false, /*reuse_b=*/false);
}

// ---------------------------------------------------------------------------
// CDNA5 async DMA: global -> LDS without touching VGPRs (ASYNCcnt-tracked).
// Guarded so the file still compiles if the toolchain lacks the builtins;
// fallback is a synchronous 16B copy with identical semantics.
// ---------------------------------------------------------------------------
#if defined(__gfx1250__) && __has_builtin(__builtin_amdgcn_global_load_async_to_lds_b128)
#define HAVE_ASYNC_LDS 1
#else
#define HAVE_ASYNC_LDS 0
#endif

__device__ __forceinline__ void async_ld_b128(const u16* gsrc, u16* ldst) {
#if HAVE_ASYNC_LDS
  __builtin_amdgcn_global_load_async_to_lds_b128(
      (AS_GLOBAL v4i*)gsrc, (AS_LOCAL v4i*)ldst, 0, 0);
#else
  *reinterpret_cast<uint4*>(ldst) = *reinterpret_cast<const uint4*>(gsrc);
#endif
}

__device__ __forceinline__ void wait_async0() {
#if defined(__gfx1250__)
#if __has_builtin(__builtin_amdgcn_s_wait_asynccnt)
  __builtin_amdgcn_s_wait_asynccnt(0);
#else
  asm volatile("s_wait_asynccnt 0x0" ::: "memory");
#endif
#endif
}

// ---------------------------------------------------------------------------
// 1) f32 -> bf16 bulk convert
// ---------------------------------------------------------------------------
__global__ void k_cvt(const float* __restrict__ in, u16* __restrict__ out, int n) {
  int i = blockIdx.x * blockDim.x + threadIdx.x;
  const int stride = gridDim.x * blockDim.x;
  for (; i < n; i += stride) out[i] = f2bf_bits(in[i]);
}

// ---------------------------------------------------------------------------
// 2) C(f32)[MxN] = A(bf16)[MxK] * B(bf16)[NxK]^T   (NT GEMM, both K-contiguous)
//    64x128 block tile, BK=32, async-to-LDS double buffering,
//    8 waves x four 16x16 accumulators (A fragment reused 4x per K-step).
// ---------------------------------------------------------------------------
#define BM 64
#define BN 128
#define BK 32

__global__ __launch_bounds__(256) void k_gemm_nt(
    const u16* __restrict__ A, const u16* __restrict__ B,
    float* __restrict__ C, int M, int N, int K) {
  __shared__ __align__(16) u16 As[2][BM][BK];   //  4 KB / buffer
  __shared__ __align__(16) u16 Bs[2][BN][BK];   //  8 KB / buffer

  const int tid  = threadIdx.x;
  const int lane = tid & 31;
  const int wave = tid >> 5;
  const int bm = blockIdx.y * BM;
  const int bn = blockIdx.x * BN;

  const int wm = (wave & 3) * 16;      // wave's 16-row strip
  const int wn = (wave >> 2) * 64;     // wave's 64-col strip (four 16x16 tiles)

  const int srow = tid >> 2;           // 0..63  : staging row
  const int scol = (tid & 3) * 8;      // 0,8,16,24 elements (16B granules)

  const int nk = K / BK;

  v8f acc[4];
  {
    v8f z = {};
#pragma unroll
    for (int i = 0; i < 4; ++i) acc[i] = z;
  }

  // stage K-tile 0 (async DMA into LDS)
  async_ld_b128(A + (size_t)(bm + srow) * K + scol,      &As[0][srow][scol]);
  async_ld_b128(B + (size_t)(bn + srow) * K + scol,      &Bs[0][srow][scol]);
  async_ld_b128(B + (size_t)(bn + 64 + srow) * K + scol, &Bs[0][64 + srow][scol]);

  const int fr   = lane & 15;
  const int koff = (lane < 16) ? 0 : 8;

  for (int kt = 0; kt < nk; ++kt) {
    const int buf = kt & 1;
    wait_async0();        // my DMA into `buf` (issued last iter / preload) is done
    __syncthreads();      // everyone's DMA into `buf` is done
    if (kt + 1 < nk) {
      const int k0 = (kt + 1) * BK;
      async_ld_b128(A + (size_t)(bm + srow) * K + k0 + scol,      &As[buf ^ 1][srow][scol]);
      async_ld_b128(B + (size_t)(bn + srow) * K + k0 + scol,      &Bs[buf ^ 1][srow][scol]);
      async_ld_b128(B + (size_t)(bn + 64 + srow) * K + k0 + scol, &Bs[buf ^ 1][64 + srow][scol]);
      if (kt + 2 < nk) {  // pull the tile after that into cache (global_prefetch_b8)
        __builtin_prefetch(A + (size_t)(bm + srow) * K + (kt + 2) * BK + scol, 0, 1);
        __builtin_prefetch(B + (size_t)(bn + srow) * K + (kt + 2) * BK + scol, 0, 1);
      }
    }
    v16bf af = ld_frag16(&As[buf][wm + fr][koff], &As[buf][wm + fr][koff + 16]);
#pragma unroll
    for (int t = 0; t < 4; ++t) {
      v16bf bf_ = ld_frag16(&Bs[buf][wn + 16 * t + fr][koff],
                            &Bs[buf][wn + 16 * t + fr][koff + 16]);
      acc[t] = wmma_bf16(af, bf_, acc[t]);
    }
  }

  // C/D layout: VGPR r, lanes 0-15 -> (m=r, n=lane); lanes 16-31 -> (m=r+8, n=lane-16)
  const int mbase = bm + wm + ((lane < 16) ? 0 : 8);
#pragma unroll
  for (int r = 0; r < 8; ++r) {
#pragma unroll
    for (int t = 0; t < 4; ++t)
      C[(size_t)(mbase + r) * N + bn + wn + 16 * t + fr] = acc[t][r];
  }
}

// ---------------------------------------------------------------------------
// 3) RMSNorm + RoPE (+ V downconvert), one block per (s, head), 64 thr = pairs
// ---------------------------------------------------------------------------
__global__ __launch_bounds__(64) void k_rmsrope(
    const float* __restrict__ qf, const float* __restrict__ kf,
    const float* __restrict__ vf,
    const float* __restrict__ nqw, const float* __restrict__ nkw,
    const float* __restrict__ freqs,           // [SEQ][HD/2][2] (cos, sin)
    u16* __restrict__ qb, u16* __restrict__ kb, u16* __restrict__ vb) {
  const int s = blockIdx.x >> 4;
  const int h = blockIdx.x & 15;
  const int t = threadIdx.x;                   // pair index 0..63
  const size_t base = ((size_t)s * HEADS + h) * HD;

  const float q0 = qf[base + 2 * t], q1 = qf[base + 2 * t + 1];
  const float k0 = kf[base + 2 * t], k1 = kf[base + 2 * t + 1];

  float sq = q0 * q0 + q1 * q1;
  float sk = k0 * k0 + k1 * k1;
#pragma unroll
  for (int o = 16; o >= 1; o >>= 1) {
    sq += __shfl_xor(sq, o, 32);
    sk += __shfl_xor(sk, o, 32);
  }
  __shared__ float red[2][2];
  if ((t & 31) == 0) { red[0][t >> 5] = sq; red[1][t >> 5] = sk; }
  __syncthreads();
  const float rq = rsqrtf((red[0][0] + red[0][1]) * (1.0f / HD) + 1e-5f);
  const float rk = rsqrtf((red[1][0] + red[1][1]) * (1.0f / HD) + 1e-5f);

  const float cs = freqs[((size_t)s * (HD / 2) + t) * 2 + 0];
  const float sn = freqs[((size_t)s * (HD / 2) + t) * 2 + 1];

  const float nq0 = q0 * rq * nqw[2 * t], nq1 = q1 * rq * nqw[2 * t + 1];
  const float nk0 = k0 * rk * nkw[2 * t], nk1 = k1 * rk * nkw[2 * t + 1];

  qb[base + 2 * t]     = f2bf_bits(nq0 * cs - nq1 * sn);
  qb[base + 2 * t + 1] = f2bf_bits(nq1 * cs + nq0 * sn);
  kb[base + 2 * t]     = f2bf_bits(nk0 * cs - nk1 * sn);
  kb[base + 2 * t + 1] = f2bf_bits(nk1 * cs + nk0 * sn);
  vb[base + 2 * t]     = f2bf_bits(vf[base + 2 * t]);
  vb[base + 2 * t + 1] = f2bf_bits(vf[base + 2 * t + 1]);
}

// ---------------------------------------------------------------------------
// 4) Flash-style SDPA. One wave per (head, 16-query tile); 32 keys / iter.
//    neg=1 remaps rows s >= NQ-L_CAP to s+L_CAP (the q2/k2/v2 substitution).
// ---------------------------------------------------------------------------
__device__ __forceinline__ int map_row(int s, int neg) {
  return (neg && s >= (NQ - L_CAP)) ? s + L_CAP : s;
}

__global__ __launch_bounds__(128) void k_attn(
    const u16* __restrict__ qb, const u16* __restrict__ kb,
    const u16* __restrict__ vb, float* __restrict__ xout, int neg) {
  __shared__ __align__(16) u16 Kt[4][32][HD];  // 32 KB : per-wave K tiles
  __shared__ __align__(16) u16 Pt[4][16][32];  //  4 KB : per-wave P staging

  const int head = blockIdx.x;
  const int wave = threadIdx.x >> 5;
  const int lane = threadIdx.x & 31;
  const int q0   = (blockIdx.y * 4 + wave) * 16;

  const int fr   = lane & 15;
  const int koff = (lane < 16) ? 0 : 8;
  const int hi8  = (lane < 16) ? 0 : 8;

  // Q fragments (A operand), resident for the whole sweep
  v16bf qfrag[4];
  {
    const int srow = map_row(q0 + fr, neg);
    const u16* qrow = qb + ((size_t)srow * HEADS + head) * HD;
#pragma unroll
    for (int c = 0; c < 4; ++c)
      qfrag[c] = ld_frag16(qrow + 32 * c + koff, qrow + 32 * c + koff + 16);
  }

  v8f acc[8];
  {
    v8f z = {};
#pragma unroll
    for (int i = 0; i < 8; ++i) acc[i] = z;
  }
  float mrow[8], lrow[8];
#pragma unroll
  for (int r = 0; r < 8; ++r) { mrow[r] = -3.0e38f; lrow[r] = 0.0f; }

  const float scale = 0.08838834764831845f;    // 1/sqrt(HD)

  for (int kt = 0; kt < NQ / 32; ++kt) {
    const int sk = kt * 32;

    // stage this wave's 32 key rows into LDS via async DMA (ASYNCcnt path)
    {
      const int krow = map_row(sk + lane, neg);
      const u16* src = kb + ((size_t)krow * HEADS + head) * HD;
      u16* dst = &Kt[wave][lane][0];
#pragma unroll
      for (int i = 0; i < HD / 8; ++i)          // 16 x b128 per lane row
        async_ld_b128(src + 8 * i, dst + 8 * i);
      if (kt + 1 < NQ / 32) {
        const int nrow = map_row(sk + 32 + lane, neg);
        __builtin_prefetch(kb + ((size_t)nrow * HEADS + head) * HD, 0, 1);
      }
    }
    wait_async0();   // same-wave consumer: only ASYNCcnt drain needed, no barrier

    // S = Q * K^T for the two 16-key sub-tiles (8 chained WMMAs)
    v8f s0 = {};
    v8f s1 = {};
#pragma unroll
    for (int c = 0; c < 4; ++c) {
      v16bf b0 = ld_frag16(&Kt[wave][fr][32 * c + koff],
                           &Kt[wave][fr][32 * c + koff + 16]);
      v16bf b1 = ld_frag16(&Kt[wave][16 + fr][32 * c + koff],
                           &Kt[wave][16 + fr][32 * c + koff + 16]);
      s0 = wmma_bf16(qfrag[c], b0, s0);
      s1 = wmma_bf16(qfrag[c], b1, s1);
    }

    // online softmax over the 32 keys (shfl reductions inside each 16-lane half)
#pragma unroll
    for (int r = 0; r < 8; ++r) {
      const float a = s0[r] * scale;
      const float b = s1[r] * scale;
      float mx = fmaxf(a, b);
#pragma unroll
      for (int o = 1; o < 16; o <<= 1) mx = fmaxf(mx, __shfl_xor(mx, o, 32));
      const float nm   = fmaxf(mrow[r], mx);
      const float corr = __expf(mrow[r] - nm);
      const float p0   = __expf(a - nm);
      const float p1   = __expf(b - nm);
      float rs = p0 + p1;
#pragma unroll
      for (int o = 1; o < 16; o <<= 1) rs += __shfl_xor(rs, o, 32);
      lrow[r] = lrow[r] * corr + rs;
      mrow[r] = nm;
#pragma unroll
      for (int t8 = 0; t8 < 8; ++t8) acc[t8][r] *= corr;
      Pt[wave][r + hi8][fr]      = f2bf_bits(p0);
      Pt[wave][r + hi8][fr + 16] = f2bf_bits(p1);
    }

    // P (16x32) back as an A fragment (LDS round-trip re-lays-out across lanes;
    // same-wave DS ops are in order, so no barrier required)
    v16bf pfrag = ld_frag16(&Pt[wave][fr][koff], &Pt[wave][fr][koff + 16]);

    // O += P * V : eight 16-wide d sub-tiles. B operand: lane%16 = d column,
    // element s = key koff + s (s<8) / 16 + koff + (s-8) (table order).
#pragma unroll
    for (int t8 = 0; t8 < 8; ++t8) {
      v16bf vfrag;
#pragma unroll
      for (int i = 0; i < 16; ++i) {
        const int vrow = map_row(sk + koff + (i < 8 ? i : i + 8), neg);
        const u16* vbase = vb + ((size_t)vrow * HEADS + head) * HD;
        vfrag[i] = __builtin_bit_cast(__bf16, vbase[16 * t8 + fr]);
      }
      acc[t8] = wmma_bf16(pfrag, vfrag, acc[t8]);
    }
  }

  // normalize and write (C/D layout as in the GEMM epilogue)
#pragma unroll
  for (int r = 0; r < 8; ++r) {
    const float inv = 1.0f / lrow[r];
    const size_t row = (size_t)(q0 + r + hi8) * DIMN + (size_t)head * HD;
#pragma unroll
    for (int t8 = 0; t8 < 8; ++t8)
      xout[row + 16 * t8 + fr] = acc[t8][r] * inv;
  }
}

// ---------------------------------------------------------------------------
// 5) NAG combine: per-row L1 norms, tau clamp, alpha blend; tail copy; bf16
// ---------------------------------------------------------------------------
__global__ __launch_bounds__(256) void k_nag(
    const float* __restrict__ xpos, const float* __restrict__ xneg,
    u16* __restrict__ xcomb) {
  const int row = blockIdx.x;                  // 0..SEQ-1
  const int tid = threadIdx.x;
  if (row >= NQ) {                             // tail rows: bf16(x_neg[:, -L:])
    const float* src = xneg + (size_t)(row - L_CAP) * DIMN;
    u16* dst = xcomb + (size_t)row * DIMN;
    for (int i = tid; i < DIMN; i += 256) dst[i] = f2bf_bits(src[i]);
    return;
  }
  const float* p = xpos + (size_t)row * DIMN;
  const float* n = xneg + (size_t)row * DIMN;
  float sp = 0.f, sg = 0.f;
  for (int i = tid; i < DIMN; i += 256) {
    const float pv = p[i], nv = n[i];
    const float g = 5.0f * pv - 4.0f * nv;     // neg*(1-5) + pos*5
    sp += fabsf(pv);
    sg += fabsf(g);
  }
#pragma unroll
  for (int o = 16; o >= 1; o >>= 1) {
    sp += __shfl_xor(sp, o, 32);
    sg += __shfl_xor(sg, o, 32);
  }
  __shared__ float rp[8], rg[8];
  if ((tid & 31) == 0) { rp[tid >> 5] = sp; rg[tid >> 5] = sg; }
  __syncthreads();
  float tp = 0.f, tg = 0.f;
#pragma unroll
  for (int i = 0; i < 8; ++i) { tp += rp[i]; tg += rg[i]; }
  float ratio = tg / tp;
  if (ratio != ratio) ratio = 10.0f;           // nan_to_num(nan=10)
  const float factor = tp * 2.5f / (tg + 1e-7f);
  const bool  cap    = ratio > 2.5f;
  u16* dst = xcomb + (size_t)row * DIMN;
  for (int i = tid; i < DIMN; i += 256) {
    const float pv = p[i], nv = n[i];
    float g = 5.0f * pv - 4.0f * nv;
    if (cap) g *= factor;
    dst[i] = f2bf_bits(g * 0.25f + pv * 0.75f);
  }
}

// ---------------------------------------------------------------------------
// Launch
// ---------------------------------------------------------------------------
extern "C" void kernel_launch(void* const* d_in, const int* in_sizes, int n_in,
                              void* d_out, int out_size, void* d_ws, size_t ws_size,
                              hipStream_t stream) {
  (void)in_sizes; (void)n_in; (void)out_size; (void)ws_size;
  const float* h   = (const float*)d_in[0];
  const float* wq  = (const float*)d_in[1];
  const float* wk  = (const float*)d_in[2];
  const float* wv  = (const float*)d_in[3];
  const float* wo  = (const float*)d_in[4];
  const float* nqw = (const float*)d_in[5];
  const float* nkw = (const float*)d_in[6];
  const float* fr  = (const float*)d_in[7];
  // d_in[8] = cap_embed_len (256), baked in as L_CAP

  char* ws = (char*)d_ws;
  size_t o = 0;
  u16* hb  = (u16*)(ws + o); o += (size_t)SEQ * DIMN * 2;
  u16* wqb = (u16*)(ws + o); o += (size_t)DIMN * DIMN * 2;
  u16* wkb = (u16*)(ws + o); o += (size_t)DIMN * DIMN * 2;
  u16* wvb = (u16*)(ws + o); o += (size_t)DIMN * DIMN * 2;
  u16* wob = (u16*)(ws + o); o += (size_t)DIMN * DIMN * 2;
  float* qf = (float*)(ws + o); o += (size_t)SEQ * DIMN * 4;
  float* kf = (float*)(ws + o); o += (size_t)SEQ * DIMN * 4;
  float* vf = (float*)(ws + o); o += (size_t)SEQ * DIMN * 4;
  u16* qb = (u16*)(ws + o); o += (size_t)SEQ * DIMN * 2;
  u16* kb = (u16*)(ws + o); o += (size_t)SEQ * DIMN * 2;
  u16* vb = (u16*)(ws + o); o += (size_t)SEQ * DIMN * 2;
  // aliases (safe: producers of qf/kf/vf are fully consumed before reuse)
  float* xpos  = qf;           // [NQ][DIMN] f32  (18.9MB <= 21MB)
  float* xneg  = kf;
  u16*   xcomb = (u16*)vf;     // [SEQ][DIMN] bf16 (10.5MB <= 21MB)

  // 1) downconvert activations + weights
  k_cvt<<<1024, 256, 0, stream>>>(h,  hb,  SEQ * DIMN);
  k_cvt<<<1024, 256, 0, stream>>>(wq, wqb, DIMN * DIMN);
  k_cvt<<<1024, 256, 0, stream>>>(wk, wkb, DIMN * DIMN);
  k_cvt<<<1024, 256, 0, stream>>>(wv, wvb, DIMN * DIMN);
  k_cvt<<<1024, 256, 0, stream>>>(wo, wob, DIMN * DIMN);

  // 2) QKV projections: [SEQ,DIMN] = hb @ W^T
  dim3 gg(DIMN / BN, SEQ / BM);
  k_gemm_nt<<<gg, 256, 0, stream>>>(hb, wqb, qf, SEQ, DIMN, DIMN);
  k_gemm_nt<<<gg, 256, 0, stream>>>(hb, wkb, kf, SEQ, DIMN, DIMN);
  k_gemm_nt<<<gg, 256, 0, stream>>>(hb, wvb, vf, SEQ, DIMN, DIMN);

  // 3) RMSNorm + RoPE + bf16 pack
  k_rmsrope<<<SEQ * HEADS, 64, 0, stream>>>(qf, kf, vf, nqw, nkw, fr, qb, kb, vb);

  // 4) the two SDPA passes (pos, and remapped "neg" guidance pass)
  dim3 ga(HEADS, NQ / 64);
  k_attn<<<ga, 128, 0, stream>>>(qb, kb, vb, xpos, 0);
  k_attn<<<ga, 128, 0, stream>>>(qb, kb, vb, xneg, 1);

  // 5) NAG combine + tail concat -> bf16
  k_nag<<<SEQ, 256, 0, stream>>>(xpos, xneg, xcomb);

  // 6) output projection straight into d_out (f32)
  k_gemm_nt<<<gg, 256, 0, stream>>>(xcomb, wob, (float*)d_out, SEQ, DIMN, DIMN);
}
